// MultiHeadAttention_36807869727333
// MI455X (gfx1250) — matrix operand order repaired and checked by reading, not compiled, not run on hardware
//
#include <hip/hip_runtime.h>

// MI455X / gfx1250 fused multi-head attention, v4.
// Phase 1: f32 -> f16 projections (Q pre-scaled by log2(e)/sqrt(Dq); V transposed).
// Phase 2: flash attention, one wave32 per (head, 32 query cols), transposed-score
//          form: T = K.Q^T, Ot = V^T.P^T. Two query tiles per wave amortize K/V
//          streaming (halves L2 traffic vs 16-col waves) -> 8 WMMA per 8 b128
//          loads per 32-KV iteration. Softmax: per-lane register reductions +
//          v_permlanex16 half swaps, raw v_exp_f32, no LDS, no barriers.
// Phase 3: f32 output projection.
//
// Workspace layout (needs 10 MB):
//   [0,2MB)  Qh  f16 [H][N][32]   (scaled by log2e/sqrt(32))
//   [2,4MB)  Kh  f16 [H][M][32]
//   [4,6MB)  Vt  f16 [H][32][M]   (transposed V)
//   [6,10MB) concat f32 [N][H*32]

#define HH  8
#define DIN 16
#define DQ  32
#define DV  32
#define NQ  4096
#define MK  4096

typedef _Float16 h2  __attribute__((ext_vector_type(2)));
typedef _Float16 h8  __attribute__((ext_vector_type(8)));
typedef _Float16 h16 __attribute__((ext_vector_type(16)));
typedef float    f8  __attribute__((ext_vector_type(8)));

// ---------------------------------------------------------------------------
// Cross-half exchange: lane i <-> lane i^16 (identity nibble selects).
// ---------------------------------------------------------------------------
__device__ __forceinline__ int xhalf_i(int x) {
#if __has_builtin(__builtin_amdgcn_permlanex16)
    return __builtin_amdgcn_permlanex16(x, x, 0x76543210, 0xFEDCBA98, false, false);
#else
    return __shfl_xor(x, 16, 32);
#endif
}
__device__ __forceinline__ float xhalf_f(float x) {
    return __builtin_bit_cast(float, xhalf_i(__builtin_bit_cast(int, x)));
}

// Raw v_exp_f32: flushes results for args < -126 to denorm/zero -- fine for
// softmax (those probabilities are ~1e-38). Avoids clang's range fixup.
__device__ __forceinline__ float fast_exp2(float x) {
#if __has_builtin(__builtin_amdgcn_exp2f)
    return __builtin_amdgcn_exp2f(x);
#else
    return exp2f(x);
#endif
}

// A-operand (16x32 f16) loader: lane row m fixed by caller's base pointer;
// element e holds column k = 16*hf + e.
__device__ __forceinline__ h16 loadA(const _Float16* __restrict__ base, int hf) {
    h8 lo = *(const h8*)(base + 16 * hf);
    h8 hi = *(const h8*)(base + 16 * hf + 8);
    h16 r;
#pragma unroll
    for (int i = 0; i < 8; ++i) { r[i] = lo[i]; r[8 + i] = hi[i]; }
    return r;
}

// ---------------------------------------------------------------------------
// Online-softmax state for one 16-query tile.
// ---------------------------------------------------------------------------
struct TileState {
    f8    o1, o2;   // Ot accumulators: reg r -> dv = r+8*hf (o1) / 16+r+8*hf (o2)
    float m, l;     // running max (log2 domain), per-lane partial sum
};

// Consume score tiles t1/t2 (reg r -> kv = r+8*hf and 16+r+8*hf within the
// 32-kv block), update state, and return the P^T B-operand for the PV WMMAs.
__device__ __forceinline__ h16 softmax_update(f8 t1, f8 t2, TileState& st, int hf) {
    // Block max over 32 kv: per-lane register max + one half-swap.
    float bm = fmaxf(t1[0], t1[1]);
#pragma unroll
    for (int r = 2; r < 8; ++r) bm = fmaxf(bm, t1[r]);
#pragma unroll
    for (int r = 0; r < 8; ++r) bm = fmaxf(bm, t2[r]);
    bm = fmaxf(bm, xhalf_f(bm));

    float mnew = fmaxf(st.m, bm);
    float corr = fast_exp2(st.m - mnew);
    st.m = mnew;

    f8 e1, e2;
#pragma unroll
    for (int r = 0; r < 8; ++r) {
        e1[r] = fast_exp2(t1[r] - mnew);
        e2[r] = fast_exp2(t2[r] - mnew);
    }

    f8 es = e1 + e2;
    float s4a = (es[0] + es[1]) + (es[2] + es[3]);
    float s4b = (es[4] + es[5]) + (es[6] + es[7]);
    st.l = st.l * corr + (s4a + s4b);
    st.o1 = st.o1 * corr;
    st.o2 = st.o2 * corr;

    // Build B-operand: element e -> kv = 16*hf + e within the block, col q = nn.
    // Lane sends what its partner needs: hf=0 sends t2-probs (kv 16..23),
    // hf=1 sends t1-probs (kv 8..15). 4 permlanex16 total.
    int ph1[4], ph2[4];
#pragma unroll
    for (int i = 0; i < 4; ++i) {
        ph1[i] = __builtin_bit_cast(int, __builtin_amdgcn_cvt_pkrtz(e1[2*i], e1[2*i+1]));
        ph2[i] = __builtin_bit_cast(int, __builtin_amdgcn_cvt_pkrtz(e2[2*i], e2[2*i+1]));
    }
    union { h16 v; int d[8]; } bp;
#pragma unroll
    for (int i = 0; i < 4; ++i) {
        int send = (hf == 0) ? ph2[i] : ph1[i];
        int recv = xhalf_i(send);
        bp.d[i]     = (hf == 0) ? ph1[i] : recv;    // kv = 16*hf     + 2i, 2i+1
        bp.d[4 + i] = (hf == 0) ? recv   : ph2[i];  // kv = 16*hf + 8 + 2i, 2i+1
    }
    return bp.v;
}

// ---------------------------------------------------------------------------
// Phase 1: per-head linear projection, f32 in, f16 out.
// ---------------------------------------------------------------------------
__global__ __launch_bounds__(256) void proj_kernel(
    const float* __restrict__ X, const float* __restrict__ W,
    const float* __restrict__ B, _Float16* __restrict__ out,
    float scale, int transpose)
{
    int tid = blockIdx.x * blockDim.x + threadIdx.x;   // H*4096*32 threads
    int c = tid & 31;
    int n = (tid >> 5) & 4095;
    int h = tid >> 17;
    float acc = B[h * 32 + c];
    const float* xr = X + (size_t)n * DIN;
    const float* wc = W + (size_t)(h * DIN) * 32 + c;
#pragma unroll
    for (int d = 0; d < DIN; ++d) acc += xr[d] * wc[d * 32];
    acc *= scale;
    size_t oidx = transpose ? ((size_t)(h * 32 + c) * 4096 + n)
                            : ((size_t)(h * 4096 + n) * 32 + c);
    out[oidx] = (_Float16)acc;
}

// ---------------------------------------------------------------------------
// Phase 2: flash attention, transposed-score form, 2 query tiles per wave.
// blockDim = 32 (one wave). grid = (N/32, H).
// ---------------------------------------------------------------------------
__global__ __launch_bounds__(32) void flash_kernel(
    const _Float16* __restrict__ Qh, const _Float16* __restrict__ Kh,
    const _Float16* __restrict__ Vt, float* __restrict__ concat)
{
    const int h     = blockIdx.y;
    const int qbase = blockIdx.x * 32;
    const int lane  = threadIdx.x;
    const int hf    = lane >> 4;      // half-wave: 0 or 1
    const int nn    = lane & 15;      // lane's query column within each tile

    // B-operands for T = K.Q^T: B[k][n] = Q[qcol][k]; element e -> k = 16*hf+e.
    h16 bqA = *(const h16*)(Qh + (size_t)(h * NQ + qbase + nn)      * DQ + 16 * hf);
    h16 bqB = *(const h16*)(Qh + (size_t)(h * NQ + qbase + 16 + nn) * DQ + 16 * hf);

    TileState A, B;
    A.o1 = f8{}; A.o2 = f8{}; A.m = -3.0e38f; A.l = 0.0f;
    B.o1 = f8{}; B.o2 = f8{}; B.m = -3.0e38f; B.l = 0.0f;

    for (int j = 0; j < MK; j += 32) {
        // A-operands shared by both query tiles.
        h16 ak1 = loadA(Kh + (size_t)(h * MK + j + nn)      * DQ, hf);
        h16 ak2 = loadA(Kh + (size_t)(h * MK + j + 16 + nn) * DQ, hf);
        h16 av1 = loadA(Vt + (size_t)(h * DV + nn)      * MK + j, hf);
        h16 av2 = loadA(Vt + (size_t)(h * DV + 16 + nn) * MK + j, hf);

        f8 z = {};
        f8 t1a = __builtin_amdgcn_wmma_f32_16x16x32_f16(false, ak1, false, bqA,
                                                        (short)0, z, false, false);
        f8 t2a = __builtin_amdgcn_wmma_f32_16x16x32_f16(false, ak2, false, bqA,
                                                        (short)0, z, false, false);
        f8 t1b = __builtin_amdgcn_wmma_f32_16x16x32_f16(false, ak1, false, bqB,
                                                        (short)0, z, false, false);
        f8 t2b = __builtin_amdgcn_wmma_f32_16x16x32_f16(false, ak2, false, bqB,
                                                        (short)0, z, false, false);

        h16 bpA = softmax_update(t1a, t2a, A, hf);
        h16 bpB = softmax_update(t1b, t2b, B, hf);

        A.o1 = __builtin_amdgcn_wmma_f32_16x16x32_f16(false, av1, false, bpA,
                                                      (short)0, A.o1, false, false);
        A.o2 = __builtin_amdgcn_wmma_f32_16x16x32_f16(false, av2, false, bpA,
                                                      (short)0, A.o2, false, false);
        B.o1 = __builtin_amdgcn_wmma_f32_16x16x32_f16(false, av1, false, bpB,
                                                      (short)0, B.o1, false, false);
        B.o2 = __builtin_amdgcn_wmma_f32_16x16x32_f16(false, av2, false, bpB,
                                                      (short)0, B.o2, false, false);
    }

    // Totals: own partial + partner partial (disjoint kv sets), then normalize.
    float invA = 1.0f / (A.l + xhalf_f(A.l));
    float invB = 1.0f / (B.l + xhalf_f(B.l));

    f8 a1 = A.o1 * invA, a2 = A.o2 * invA;
    f8 b1 = B.o1 * invB, b2 = B.o2 * invB;

    // Ot C-layout: o1 reg r -> dv = r + 8*hf ; o2 reg r -> dv = 16 + r + 8*hf.
    float* orowA = concat + (size_t)(qbase + nn) * (HH * DV) + h * DV;
    *(f8*)(orowA + 8 * hf)      = a1;
    *(f8*)(orowA + 16 + 8 * hf) = a2;
    float* orowB = concat + (size_t)(qbase + 16 + nn) * (HH * DV) + h * DV;
    *(f8*)(orowB + 8 * hf)      = b1;
    *(f8*)(orowB + 16 + 8 * hf) = b2;
}

// ---------------------------------------------------------------------------
// Phase 3: out[n][e] = concat[n][:] . Wo[:, e] + bo[e]   (f32)
// ---------------------------------------------------------------------------
__global__ __launch_bounds__(256) void out_proj_kernel(
    const float* __restrict__ concat, const float* __restrict__ Wo,
    const float* __restrict__ bo, float* __restrict__ out)
{
    int tid = blockIdx.x * blockDim.x + threadIdx.x;   // 4096*16 threads
    int e = tid & 15;
    int n = tid >> 4;
    float acc = bo[e];
    const float* cr = concat + (size_t)n * (HH * DV);
#pragma unroll 8
    for (int c = 0; c < HH * DV; ++c) acc += cr[c] * Wo[c * DIN + e];
    out[tid] = acc;
}

// ---------------------------------------------------------------------------
extern "C" void kernel_launch(void* const* d_in, const int* in_sizes, int n_in,
                              void* d_out, int out_size, void* d_ws, size_t ws_size,
                              hipStream_t stream)
{
    const float* x  = (const float*)d_in[0];
    const float* y  = (const float*)d_in[1];
    const float* Wq = (const float*)d_in[2];
    const float* bq = (const float*)d_in[3];
    const float* Wk = (const float*)d_in[4];
    const float* bk = (const float*)d_in[5];
    const float* Wv = (const float*)d_in[6];
    const float* bv = (const float*)d_in[7];
    const float* Wo = (const float*)d_in[8];
    const float* bo = (const float*)d_in[9];
    float* out = (float*)d_out;

    char* ws = (char*)d_ws;
    _Float16* Qh = (_Float16*)(ws);
    _Float16* Kh = (_Float16*)(ws + ((size_t)2 << 20));
    _Float16* Vt = (_Float16*)(ws + ((size_t)4 << 20));
    float* concat = (float*)(ws + ((size_t)6 << 20));

    const int projThreads = HH * 4096 * 32;            // 1,048,576
    // Q scale = log2(e) / sqrt(D_Q): softmax done in exp2 domain.
    const float qscale = 1.4426950408889634f * 0.17677669529663687f;
    proj_kernel<<<projThreads / 256, 256, 0, stream>>>(x, Wq, bq, Qh, qscale, 0);
    proj_kernel<<<projThreads / 256, 256, 0, stream>>>(y, Wk, bk, Kh, 1.0f, 0);
    proj_kernel<<<projThreads / 256, 256, 0, stream>>>(y, Wv, bv, Vt, 1.0f, 1);

    flash_kernel<<<dim3(NQ / 32, HH), 32, 0, stream>>>(Qh, Kh, Vt, concat);

    out_proj_kernel<<<(NQ * DIN) / 256, 256, 0, stream>>>(concat, Wo, bo, out);
}
